// MultiHeadAttention_82437602279490
// MI455X (gfx1250) — matrix operand order repaired and checked by reading, not compiled, not run on hardware
//
#include <hip/hip_runtime.h>
#include <hip/hip_bf16.h>
#include <stdint.h>

// ---------------------------------------------------------------------------
// MHA block for MI455X (gfx1250, wave32, WMMA, TDM).
//   out = LayerNorm( softmax(QK^T/sqrt(dh)) V @ Wo^T + q )
// f16 data / f32 accumulate via v_wmma_f32_16x16x32_f16.
// f32->f16 conversion fused into GEMM LDS staging (no separate passes).
// Flash-attention K tiles prefetched by the Tensor Data Mover (guarded).
// ---------------------------------------------------------------------------

typedef _Float16 half_t;
typedef __attribute__((ext_vector_type(8)))  _Float16 v8h;
typedef __attribute__((ext_vector_type(16))) _Float16 v16h;
typedef __attribute__((ext_vector_type(8)))  float    v8f;
typedef __attribute__((ext_vector_type(4)))  float    v4f;
typedef __attribute__((ext_vector_type(4)))  unsigned int u32x4;
typedef __attribute__((ext_vector_type(4)))  int      i32x4;
typedef __attribute__((ext_vector_type(8)))  int      i32x8;

#define D_MODEL 1024
#define SEQ_N   2048
#define BATCH   4
#define NHEAD   16
#define DH      64
#define M_TOTAL (BATCH * SEQ_N)   // 8192 rows

// Tensor Data Mover availability (device pass only; host pass parses fallback)
#ifdef __has_builtin
#if __has_builtin(__builtin_amdgcn_tensor_load_to_lds) && \
    __has_builtin(__builtin_amdgcn_s_wait_tensorcnt)
#define HAS_TDM 1
#endif
#endif
#ifndef HAS_TDM
#define HAS_TDM 0
#endif

__device__ __forceinline__ v16h cat8(v8h lo, v8h hi) {
  v16h r;
#pragma unroll
  for (int i = 0; i < 8; ++i) { r[i] = lo[i]; r[i + 8] = hi[i]; }
  return r;
}

__device__ __forceinline__ v8f vzero8() {
  v8f z = {0.f, 0.f, 0.f, 0.f, 0.f, 0.f, 0.f, 0.f};
  return z;
}

// 16 contiguous elements -> two v8h, converting f32 -> f16 when needed.
__device__ __forceinline__ void ldcvt16(const half_t* __restrict__ g, v8h& lo, v8h& hi) {
  lo = *(const v8h*)g;
  hi = *(const v8h*)(g + 8);
}
__device__ __forceinline__ void ldcvt16(const float* __restrict__ g, v8h& lo, v8h& hi) {
  v4f f0 = *(const v4f*)(g);
  v4f f1 = *(const v4f*)(g + 4);
  v4f f2 = *(const v4f*)(g + 8);
  v4f f3 = *(const v4f*)(g + 12);
#pragma unroll
  for (int i = 0; i < 4; ++i) {
    lo[i]     = (half_t)f0[i];
    lo[i + 4] = (half_t)f1[i];
    hi[i]     = (half_t)f2[i];
    hi[i + 4] = (half_t)f3[i];
  }
}

#if HAS_TDM
// TDM: load a 64x64 f16 tile (row stride 1024 elements) into LDS with +16B
// padding per 128B row (LDS row stride 144B = 72 halves). One issue per wave.
__device__ __forceinline__ void tdm_load_tile64(unsigned lds_off, const half_t* gptr) {
  unsigned long long ga = (unsigned long long)(uintptr_t)gptr;
  u32x4 g0;
  g0[0] = 1u;                                   // count=1 (valid user descriptor)
  g0[1] = lds_off;                              // lds_addr (bytes)
  g0[2] = (unsigned)(ga & 0xffffffffu);         // global_addr[31:0]
  g0[3] = (unsigned)((ga >> 32) & 0x1ffffffu)   // global_addr[56:32]
        | (2u << 30);                           // type = 2 ("image")
  // group1 bitfields (256b), built as 4x u64:
  //  [17:16] data_size=1 (2B)   [20] pad_enable=1
  //  [24:22] pad_interval=4 (32 DWORDs = 128B)  [31:25] pad_amount=3 (4 DWORDs = 16B)
  //  [79:48] tensor_dim0=64     [111:80] tensor_dim1=64
  //  [127:112] tile_dim0=64     [143:128] tile_dim1=64
  //  [207:160] tensor_dim0_stride=1024
  unsigned long long q0 = (unsigned long long)((1u << 16) | (1u << 20) |
                                               (4u << 22) | (3u << 25)) |
                          (64ull << 48);
  unsigned long long q1 = (64ull << 16) | (64ull << 48);
  unsigned long long q2 = 64ull | (1024ull << 32);
  unsigned long long q3 = 0ull;
  i32x8 g1;
  g1[0] = (int)q0; g1[1] = (int)(q0 >> 32);
  g1[2] = (int)q1; g1[3] = (int)(q1 >> 32);
  g1[4] = (int)q2; g1[5] = (int)(q2 >> 32);
  g1[6] = (int)q3; g1[7] = (int)(q3 >> 32);
  i32x4 z4 = {0, 0, 0, 0};
#if __clang_major__ >= 23
  i32x8 z8 = {0, 0, 0, 0, 0, 0, 0, 0};
  __builtin_amdgcn_tensor_load_to_lds(g0, g1, z4, z4, z8, 0);
#else
  __builtin_amdgcn_tensor_load_to_lds(g0, g1, z4, z4, 0);
#endif
}
#endif  // HAS_TDM

// ---------------------------------------------------------------------------
// C[M,1024] = scale * (X[M,1024] @ W[1024,1024]^T)   (torch Linear convention)
// f32 or f16 inputs (converted to f16 during LDS staging). 256 thr = 8 waves;
// block tile 128x128; K-step 32; double-buffered LDS, 1 barrier per stage.
// ---------------------------------------------------------------------------
template <typename XT, typename WT, typename OutT>
__global__ __launch_bounds__(256)
void gemm_xwt(const XT* __restrict__ X, const WT* __restrict__ W,
              OutT* __restrict__ out, float scale) {
  __shared__ half_t As[2][128][40];   // 40 = 32 + 8 pad (bank-conflict break)
  __shared__ half_t Bs[2][128][40];

  const int tid  = threadIdx.x;
  const int lane = tid & 31;
  const int wave = tid >> 5;
  const int wm   = wave & 3;   // 0..3 : 32-row slab
  const int wn   = wave >> 2;  // 0..1 : 64-col slab
  const int m0   = blockIdx.y * 128;
  const int n0   = blockIdx.x * 128;
  const int lrow = lane & 15;
  const int hsel = lane >> 4;

  v8f acc[2][4];
#pragma unroll
  for (int i = 0; i < 2; ++i)
#pragma unroll
    for (int j = 0; j < 4; ++j) acc[i][j] = vzero8();

  const int ldrow = tid >> 1;          // 0..127
  const int ldseg = (tid & 1) * 16;    // 0 or 16 halves
  const XT* gx = X + (long)(m0 + ldrow) * D_MODEL + ldseg;
  const WT* gw = W + (long)(n0 + ldrow) * D_MODEL + ldseg;

  // stage 0
  v8h a0, a1, b0, b1;
  ldcvt16(gx, a0, a1);
  ldcvt16(gw, b0, b1);
  *(v8h*)&As[0][ldrow][ldseg]     = a0;
  *(v8h*)&As[0][ldrow][ldseg + 8] = a1;
  *(v8h*)&Bs[0][ldrow][ldseg]     = b0;
  *(v8h*)&Bs[0][ldrow][ldseg + 8] = b1;
  __syncthreads();

  const int NST = D_MODEL / 32;
  for (int s = 0; s < NST; ++s) {
    const int cur = s & 1;
    const bool more = (s + 1 < NST);
    if (more) {
      ldcvt16(gx + (s + 1) * 32, a0, a1);   // prefetch next stage into regs
      ldcvt16(gw + (s + 1) * 32, b0, b1);
      if (s + 2 < NST) {
        __builtin_prefetch(gx + (s + 2) * 32, 0, 0);
        __builtin_prefetch(gw + (s + 2) * 32, 0, 0);
      }
    }

    // A fragments: 16x32, lane holds row (lane&15); ISA 16-bit A K-map
    v16h afr[2];
#pragma unroll
    for (int tm = 0; tm < 2; ++tm) {
      const half_t* ar = &As[cur][wm * 32 + tm * 16 + lrow][0];
      v8h lo = *(const v8h*)(ar + hsel * 8);
      v8h hi = *(const v8h*)(ar + 16 + hsel * 8);
      afr[tm] = cat8(lo, hi);
    }
    // B fragments: 32x16, lane holds col (lane&15); same K-map as A
    v16h bfr[4];
#pragma unroll
    for (int tn = 0; tn < 4; ++tn) {
      const half_t* br = &Bs[cur][wn * 64 + tn * 16 + lrow][0];
      v8h lo = *(const v8h*)(br + hsel * 8);
      v8h hi = *(const v8h*)(br + 16 + hsel * 8);
      bfr[tn] = cat8(lo, hi);
    }
#pragma unroll
    for (int tm = 0; tm < 2; ++tm)
#pragma unroll
      for (int tn = 0; tn < 4; ++tn)
        acc[tm][tn] = __builtin_amdgcn_wmma_f32_16x16x32_f16(
            false, afr[tm], false, bfr[tn], (short)0, acc[tm][tn], false, false);

    if (more) {
      const int nxt = cur ^ 1;
      *(v8h*)&As[nxt][ldrow][ldseg]     = a0;
      *(v8h*)&As[nxt][ldrow][ldseg + 8] = a1;
      *(v8h*)&Bs[nxt][ldrow][ldseg]     = b0;
      *(v8h*)&Bs[nxt][ldrow][ldseg + 8] = b1;
    }
    __syncthreads();
  }

  // epilogue: D layout row = r + hsel*8, col = lane&15
#pragma unroll
  for (int tm = 0; tm < 2; ++tm)
#pragma unroll
    for (int tn = 0; tn < 4; ++tn)
#pragma unroll
      for (int r = 0; r < 8; ++r) {
        int row = m0 + wm * 32 + tm * 16 + r + hsel * 8;
        int col = n0 + wn * 64 + tn * 16 + lrow;
        float v = acc[tm][tn][r] * scale;
        out[(long)row * D_MODEL + col] = (OutT)v;
      }
}

// ---------------------------------------------------------------------------
// Streaming-softmax attention. Grid: x = N/64 query blocks, y = B*H.
// 128 threads = 4 waves; wave w owns query rows [w*16, w*16+16).
// K tiles: TDM double-buffered prefetch (DMA overlaps compute) when available.
// Q pre-scaled by 1/sqrt(Dh) at projection time.
// ---------------------------------------------------------------------------
__global__ __launch_bounds__(128)
void flash_attn(const half_t* __restrict__ Qp, const half_t* __restrict__ Kp,
                const half_t* __restrict__ Vp, half_t* __restrict__ AO) {
  __shared__ half_t Ks[2][64][72];     // keys row-major [key][dh], double buffer
  __shared__ half_t Vt[64][72];        // values transposed [dh][key]
  __shared__ half_t Ps[4][16][72];     // per-wave P tile relay (D -> A layout)

  const int tid  = threadIdx.x;
  const int lane = tid & 31;
  const int wave = tid >> 5;
  const int b    = blockIdx.y >> 4;
  const int h    = blockIdx.y & 15;
  const int q0   = blockIdx.x * 64;
  const int lrow = lane & 15;
  const int hsel = lane >> 4;

  // Q fragments (A operand), two 32-wide K chunks over Dh=64
  const long qbase = ((long)(b * SEQ_N + q0 + wave * 16 + lrow)) * D_MODEL + h * DH;
  v16h qa[2];
#pragma unroll
  for (int c = 0; c < 2; ++c) {
    v8h lo = *(const v8h*)(Qp + qbase + c * 32 + hsel * 8);
    v8h hi = *(const v8h*)(Qp + qbase + c * 32 + 16 + hsel * 8);
    qa[c] = cat8(lo, hi);
  }

  v8f o[4];
#pragma unroll
  for (int tn = 0; tn < 4; ++tn) o[tn] = vzero8();
  float m[8], l[8];
#pragma unroll
  for (int r = 0; r < 8; ++r) { m[r] = -1e30f; l[r] = 0.f; }

  const long kvbase = ((long)(b * SEQ_N)) * D_MODEL + h * DH;
  const int NKB = SEQ_N / 64;

#if HAS_TDM
  if (wave == 0)
    tdm_load_tile64((unsigned)(uintptr_t)(void*)&Ks[0][0][0], Kp + kvbase);
#endif

  for (int s = 0; s < NKB; ++s) {
    const int j = s * 64;
    const int cur = s & 1;

    // --- stage V (transposed) manually; K via TDM (or fallback loads) ---
#pragma unroll
    for (int it = 0; it < 2; ++it) {
      int c   = tid + it * 128;        // 256 chunks of 16 halves
      int row = c >> 2;
      int seg = (c & 3) * 16;
      long g  = kvbase + (long)(j + row) * D_MODEL + seg;
      v8h v0v = *(const v8h*)(Vp + g);
      v8h v1v = *(const v8h*)(Vp + g + 8);
#pragma unroll
      for (int e = 0; e < 8; ++e) {
        Vt[seg + e][row]     = v0v[e];
        Vt[seg + 8 + e][row] = v1v[e];
      }
#if !HAS_TDM
      v8h k0v = *(const v8h*)(Kp + g);
      v8h k1v = *(const v8h*)(Kp + g + 8);
      *(v8h*)&Ks[cur][row][seg]     = k0v;
      *(v8h*)&Ks[cur][row][seg + 8] = k1v;
#endif
    }
#if HAS_TDM
    if (wave == 0) {
      if (s + 1 < NKB) {
        tdm_load_tile64((unsigned)(uintptr_t)(void*)&Ks[cur ^ 1][0][0],
                        Kp + kvbase + (long)(j + 64) * D_MODEL);
        __builtin_amdgcn_s_wait_tensorcnt(1);   // oldest (current tile) done
      } else {
        __builtin_amdgcn_s_wait_tensorcnt(0);
      }
    }
#endif
    __syncthreads();

    // --- S = Q K^T : 4 col-tiles x 2 K-chunks of WMMA ---
    v8f st[4];
#pragma unroll
    for (int tn = 0; tn < 4; ++tn) st[tn] = vzero8();
#pragma unroll
    for (int c = 0; c < 2; ++c) {
#pragma unroll
      for (int tn = 0; tn < 4; ++tn) {
        const half_t* kr = &Ks[cur][tn * 16 + lrow][c * 32 + hsel * 8];
        v8h lo = *(const v8h*)kr;
        v8h hi = *(const v8h*)(kr + 16);
        v16h kb = cat8(lo, hi);
        st[tn] = __builtin_amdgcn_wmma_f32_16x16x32_f16(
            false, qa[c], false, kb, (short)0, st[tn], false, false);
      }
    }

    // --- online softmax per owned row (row = r + hsel*8) ---
#pragma unroll
    for (int r = 0; r < 8; ++r) {
      float mx = -1e30f;
#pragma unroll
      for (int tn = 0; tn < 4; ++tn) mx = fmaxf(mx, st[tn][r]);
#pragma unroll
      for (int msk = 1; msk < 16; msk <<= 1) mx = fmaxf(mx, __shfl_xor(mx, msk, 32));
      float nm    = fmaxf(m[r], mx);
      float alpha = __expf(m[r] - nm);
      m[r] = nm;
      float rs = 0.f;
#pragma unroll
      for (int tn = 0; tn < 4; ++tn) {
        float p = __expf(st[tn][r] - nm);
        st[tn][r] = p;
        rs += p;
      }
#pragma unroll
      for (int msk = 1; msk < 16; msk <<= 1) rs += __shfl_xor(rs, msk, 32);
      l[r] = l[r] * alpha + rs;
#pragma unroll
      for (int tn = 0; tn < 4; ++tn) o[tn][r] *= alpha;
#pragma unroll
      for (int tn = 0; tn < 4; ++tn)
        Ps[wave][r + hsel * 8][tn * 16 + lrow] = (half_t)st[tn][r];
    }

    // --- O += P V : A from Ps (wave-local), B from Vt ---
#pragma unroll
    for (int c = 0; c < 2; ++c) {
      const half_t* pr = &Ps[wave][lrow][c * 32];
      v8h plo = *(const v8h*)(pr + hsel * 8);
      v8h phi = *(const v8h*)(pr + 16 + hsel * 8);
      v16h pa = cat8(plo, phi);
#pragma unroll
      for (int tn = 0; tn < 4; ++tn) {
        const half_t* vr = &Vt[tn * 16 + lrow][c * 32 + hsel * 8];
        v8h lo = *(const v8h*)vr;
        v8h hi = *(const v8h*)(vr + 16);
        v16h vb = cat8(lo, hi);
        o[tn] = __builtin_amdgcn_wmma_f32_16x16x32_f16(
            false, pa, false, vb, (short)0, o[tn], false, false);
      }
    }
    __syncthreads();
  }

  // --- normalize and write merged-head layout [B*N, D], col = h*64+dh ---
#pragma unroll
  for (int r = 0; r < 8; ++r) {
    float inv = 1.0f / l[r];
    int qrow = q0 + wave * 16 + r + hsel * 8;
    long obase = ((long)(b * SEQ_N + qrow)) * D_MODEL + h * DH;
#pragma unroll
    for (int tn = 0; tn < 4; ++tn)
      AO[obase + tn * 16 + lrow] = (half_t)(o[tn][r] * inv);
  }
}

// ---------------------------------------------------------------------------
// out = LayerNorm(tmp + q) * gamma + beta ; one block per row
// ---------------------------------------------------------------------------
__global__ __launch_bounds__(256)
void residual_layernorm(const float* __restrict__ tmp, const float* __restrict__ qres,
                        const float* __restrict__ gamma, const float* __restrict__ beta,
                        float* __restrict__ out) {
  __shared__ float xs[D_MODEL];
  __shared__ float red[16];
  const int row = blockIdx.x;
  const long base = (long)row * D_MODEL;
  float s = 0.f, s2 = 0.f;
  for (int c = threadIdx.x; c < D_MODEL; c += 256) {
    float x = tmp[base + c] + qres[base + c];
    xs[c] = x;
    s += x;
    s2 += x * x;
  }
#pragma unroll
  for (int msk = 1; msk < 32; msk <<= 1) {
    s  += __shfl_xor(s, msk, 32);
    s2 += __shfl_xor(s2, msk, 32);
  }
  const int wave = threadIdx.x >> 5, lane = threadIdx.x & 31;
  if (lane == 0) { red[wave] = s; red[8 + wave] = s2; }
  __syncthreads();
  if (threadIdx.x == 0) {
    float ts = 0.f, t2 = 0.f;
    for (int i = 0; i < 8; ++i) { ts += red[i]; t2 += red[8 + i]; }
    red[0] = ts;
    red[8] = t2;
  }
  __syncthreads();
  const float mu  = red[0] / (float)D_MODEL;
  const float var = red[8] / (float)D_MODEL - mu * mu;
  const float rsd = rsqrtf(var + 1e-5f);
  for (int c = threadIdx.x; c < D_MODEL; c += 256)
    out[base + c] = (xs[c] - mu) * rsd * gamma[c] + beta[c];
}

// ---------------------------------------------------------------------------
// Host launcher. Inputs: q,k,v,Wq,Wk,Wv,Wo,gamma,beta (all f32). Output f32.
// ---------------------------------------------------------------------------
extern "C" void kernel_launch(void* const* d_in, const int* in_sizes, int n_in,
                              void* d_out, int out_size, void* d_ws, size_t ws_size,
                              hipStream_t stream) {
  (void)in_sizes; (void)n_in; (void)out_size; (void)ws_size;

  const float* q     = (const float*)d_in[0];
  const float* k     = (const float*)d_in[1];
  const float* v     = (const float*)d_in[2];
  const float* Wq    = (const float*)d_in[3];
  const float* Wk    = (const float*)d_in[4];
  const float* Wv    = (const float*)d_in[5];
  const float* Wo    = (const float*)d_in[6];
  const float* gamma = (const float*)d_in[7];
  const float* beta  = (const float*)d_in[8];
  float* out = (float*)d_out;

  const size_t MB = 1ull << 20;
  char* ws = (char*)d_ws;
  half_t* Qp  = (half_t*)(ws + 0 * MB);    // 16 MB : projected Q (pre-scaled)
  half_t* Kp  = (half_t*)(ws + 16 * MB);   // 16 MB
  half_t* Vp  = (half_t*)(ws + 32 * MB);   // 16 MB
  half_t* AO  = (half_t*)(ws + 48 * MB);   // 16 MB : attention out, merged heads
  float*  tmp = (float*)(ws + 64 * MB);    // 32 MB : output projection (f32)

  // QKV projections, f32 inputs converted in-kernel (Q folds in 1/sqrt(Dh))
  dim3 ggrid(D_MODEL / 128, M_TOTAL / 128);  // (8, 64)
  gemm_xwt<float, float, half_t><<<ggrid, 256, 0, stream>>>(q, Wq, Qp, 0.125f);
  gemm_xwt<float, float, half_t><<<ggrid, 256, 0, stream>>>(k, Wk, Kp, 1.0f);
  gemm_xwt<float, float, half_t><<<ggrid, 256, 0, stream>>>(v, Wv, Vp, 1.0f);

  // attention
  flash_attn<<<dim3(SEQ_N / 64, BATCH * NHEAD), 128, 0, stream>>>(Qp, Kp, Vp, AO);

  // output projection (f16 activations x f32 weights -> f32)
  gemm_xwt<half_t, float, float><<<ggrid, 256, 0, stream>>>(AO, Wo, tmp, 1.0f);

  // residual + LayerNorm
  residual_layernorm<<<M_TOTAL, 256, 0, stream>>>(tmp, q, gamma, beta, out);
}
